// SwinTransformerBlock_58875411694110
// MI455X (gfx1250) — compile-verified, matrix operand order checked by reading, not run on hardware
//
#include <hip/hip_runtime.h>
#include <stdint.h>

// ---------------- problem constants ----------------
#define DIMC   768
#define HEADS  48
#define HD     16
#define WS     3
#define TSEQ   4096
#define BATCH  8
#define TP     4098              // TSEQ padded to multiple of WS
#define NWIN   (TP / WS)         // 1366
#define EPSLN  1e-5f

// ---------------- WMMA types ----------------
typedef __bf16 bf16_t;
typedef bf16_t v16bf __attribute__((ext_vector_type(16)));
typedef float  v8f   __attribute__((ext_vector_type(8)));

struct __attribute__((aligned(16))) U4 { uint32_t x, y, z, w; };
union Frag { U4 q[2]; v16bf v; };   // 32 bytes

// ---------------- helpers ----------------
__device__ __forceinline__ float bf2f(uint16_t h) {
    return __uint_as_float(((uint32_t)h) << 16);
}
__device__ __forceinline__ uint16_t f2bf(float f) {
    uint32_t u = __float_as_uint(f);
    uint32_t r = u + 0x7fffu + ((u >> 16) & 1u);   // round-to-nearest-even
    return (uint16_t)(r >> 16);
}
__device__ __forceinline__ uint32_t lds_addr_of(const void* p) {
    // flat LDS addresses carry the LDS byte offset in the low 32 bits
    return (uint32_t)(uintptr_t)p;
}

// ---------------- weight convert + transpose: f32 [K][N] -> bf16 [N][K] ----------------
__global__ __launch_bounds__(256) void wt_transpose_bf16(
    const float* __restrict__ W, uint16_t* __restrict__ Wt, int K, int N)
{
    int idx = blockIdx.x * 256 + threadIdx.x;
    if (idx >= K * N) return;
    int k = idx / N, n = idx % N;
    Wt[(size_t)n * K + k] = f2bf(W[idx]);
}

// ---------------- fused LayerNorm -> bf16 (optionally rolled/padded layout) ----------------
__global__ __launch_bounds__(256) void ln_bf16_kernel(
    const float* __restrict__ x, const float* __restrict__ g,
    const float* __restrict__ b, uint16_t* __restrict__ out, int rollmode)
{
    __shared__ float red[256];
    __shared__ float red2[256];
    const int token = blockIdx.x;                 // 0 .. BATCH*TSEQ-1
    const int tid   = threadIdx.x;
    const float* xr = x + (size_t)token * DIMC;

    const int i0 = tid * 3;
    float v0 = xr[i0 + 0], v1 = xr[i0 + 1], v2 = xr[i0 + 2];
    red[tid]  = v0 + v1 + v2;
    red2[tid] = v0 * v0 + v1 * v1 + v2 * v2;
    __syncthreads();
#pragma unroll
    for (int s = 128; s > 0; s >>= 1) {
        if (tid < s) { red[tid] += red[tid + s]; red2[tid] += red2[tid + s]; }
        __syncthreads();
    }
    const float mean = red[0] * (1.0f / DIMC);
    const float var  = red2[0] * (1.0f / DIMC) - mean * mean;
    const float rs   = rsqrtf(var + EPSLN);

    size_t orow;
    if (rollmode) {                                // xs[dst] = LN(x[t]), dst = (t-1) mod TP
        int bb = token / TSEQ, t = token % TSEQ;
        int dst = (t == 0) ? (TP - 1) : (t - 1);
        orow = ((size_t)bb * TP + dst) * DIMC;
    } else {
        orow = (size_t)token * DIMC;
    }
    out[orow + i0 + 0] = f2bf((v0 - mean) * rs * g[i0 + 0] + b[i0 + 0]);
    out[orow + i0 + 1] = f2bf((v1 - mean) * rs * g[i0 + 1] + b[i0 + 1]);
    out[orow + i0 + 2] = f2bf((v2 - mean) * rs * g[i0 + 2] + b[i0 + 2]);
}

// zero the two padding rows (t = TSEQ-1+{0,1} in rolled space) of xs per batch
__global__ __launch_bounds__(256) void zero_pad_rows(uint16_t* __restrict__ xs)
{
    int blk   = blockIdx.x;           // BATCH*2 blocks
    int bb    = blk >> 1;
    int which = blk & 1;              // row TSEQ-1 or TSEQ
    size_t row = ((size_t)bb * TP + (TSEQ - 1 + which)) * DIMC;
    int i0 = threadIdx.x * 3;
    xs[row + i0 + 0] = 0;
    xs[row + i0 + 1] = 0;
    xs[row + i0 + 2] = 0;
}

// ---------------- WMMA GEMM with async-to-LDS double buffering ----------------
// C[M,N] = A[M,K](bf16, row major) * Bt[N,K](bf16, weight pre-transposed)
// modes: 0 = store bf16
//        1 = roll(+1)/truncate + residual(x) -> f32  (attention proj epilogue)
//        2 = +bias, ReLU -> bf16                      (mlp1)
//        3 = +bias + residual -> f32                  (mlp2, writes d_out)
#define BM 128
#define BN 128
#define BK 32
#define LDT 40    // padded LDS row stride (elements) to spread banks

__global__ __launch_bounds__(256) void gemm_wmma_bf16(
    const uint16_t* __restrict__ A, const uint16_t* __restrict__ Bt,
    int M, int N, int K, int mode,
    const float* __restrict__ bias, const float* __restrict__ resid,
    void* __restrict__ Out)
{
    __shared__ __align__(16) uint16_t As[2][BM * LDT];
    __shared__ __align__(16) uint16_t Bs[2][BN * LDT];

    const int tid  = threadIdx.x;
    const int lane = tid & 31;
    const int wave = tid >> 5;       // 0..7
    const int wm   = wave >> 1;      // 0..3 : 32-row strip
    const int wn   = wave & 1;       // 0..1 : 64-col strip
    const int l15  = lane & 15;
    const int half = lane >> 4;

    const int m0 = blockIdx.y * BM;
    const int n0 = blockIdx.x * BN;

    v8f acc[2][4];
#pragma unroll
    for (int i = 0; i < 2; ++i)
#pragma unroll
        for (int j = 0; j < 4; ++j)
#pragma unroll
            for (int e = 0; e < 8; ++e) acc[i][j][e] = 0.0f;

    const int row = tid >> 1;        // 0..127
    const int seg = tid & 1;         // two 16-element halves of a 32-wide K tile

    // clamped global rows: keeps async issue count uniform per wave (ASYNCcnt
    // bookkeeping must not depend on EXEC); garbage rows are masked at store.
    int gmA = m0 + row; if (gmA >= M) gmA = 0;
    int gnB = n0 + row; if (gnB >= N) gnB = 0;
    const uint16_t* gA = A  + (size_t)gmA * K + seg * 16;
    const uint16_t* gB = Bt + (size_t)gnB * K + seg * 16;
    const uint32_t sAoff = (uint32_t)(row * LDT + seg * 16);
    const uint32_t sBoff = sAoff;

    const int KT = K / BK;           // K is always a multiple of 32 here

    // stage issuer: 4 async b128 loads per thread (A:2, B:2), ASYNCcnt += 4 per wave
    auto issue_stage = [&](int kt, int buf) {
        const uint64_t ga = (uint64_t)(uintptr_t)(gA + (size_t)kt * BK);
        const uint64_t gb = (uint64_t)(uintptr_t)(gB + (size_t)kt * BK);
        const uint32_t la = lds_addr_of(&As[buf][sAoff]);
        const uint32_t lb = lds_addr_of(&Bs[buf][sBoff]);
        asm volatile("global_load_async_to_lds_b128 %0, %1, off"
                     :: "v"(la), "v"(ga) : "memory");
        asm volatile("global_load_async_to_lds_b128 %0, %1, off offset:16"
                     :: "v"(la + 16u), "v"(ga) : "memory");
        asm volatile("global_load_async_to_lds_b128 %0, %1, off"
                     :: "v"(lb), "v"(gb) : "memory");
        asm volatile("global_load_async_to_lds_b128 %0, %1, off offset:16"
                     :: "v"(lb + 16u), "v"(gb) : "memory");
    };

    issue_stage(0, 0);

    for (int kt = 0; kt < KT; ++kt) {
        const int buf = kt & 1;
        if (kt + 1 < KT) {
            issue_stage(kt + 1, buf ^ 1);
            // retire stage kt (async loads complete in order): ASYNCcnt 8 -> 4
            asm volatile("s_wait_asynccnt 0x4" ::: "memory");
        } else {
            asm volatile("s_wait_asynccnt 0x0" ::: "memory");
        }
        __syncthreads();   // all waves' tile portions landed in LDS[buf]

        // ---- fragments (ISA 16-bit layout: lane half selects K{0..7,16..23}/{8..15,24..31}) ----
        Frag aF[2], bF[4];
#pragma unroll
        for (int mi = 0; mi < 2; ++mi) {
            int ab = (wm * 32 + mi * 16 + l15) * LDT + half * 8;
            aF[mi].q[0] = *(const U4*)&As[buf][ab];
            aF[mi].q[1] = *(const U4*)&As[buf][ab + 16];
        }
#pragma unroll
        for (int nj = 0; nj < 4; ++nj) {
            int bb = (wn * 64 + nj * 16 + l15) * LDT + half * 8;
            bF[nj].q[0] = *(const U4*)&Bs[buf][bb];
            bF[nj].q[1] = *(const U4*)&Bs[buf][bb + 16];
        }
#pragma unroll
        for (int mi = 0; mi < 2; ++mi)
#pragma unroll
            for (int nj = 0; nj < 4; ++nj)
                acc[mi][nj] = __builtin_amdgcn_wmma_f32_16x16x32_bf16(
                    false, aF[mi].v, false, bF[nj].v,
                    (short)0, acc[mi][nj], false, false);

        __syncthreads();   // LDS[buf] free for reuse at stage kt+2
    }

    // ---- epilogue ----
#pragma unroll
    for (int mi = 0; mi < 2; ++mi) {
#pragma unroll
        for (int nj = 0; nj < 4; ++nj) {
            const int mbase = m0 + wm * 32 + mi * 16 + half * 8;
            const int n     = n0 + wn * 64 + nj * 16 + l15;
            if (n >= N) continue;
#pragma unroll
            for (int r = 0; r < 8; ++r) {
                const int m = mbase + r;
                if (m >= M) continue;
                float v = acc[mi][nj][r];
                if (mode == 0) {
                    ((uint16_t*)Out)[(size_t)m * N + n] = f2bf(v);
                } else if (mode == 1) {
                    // row m lives in rolled space: final token = (t+1) mod TP, keep < TSEQ
                    int bb = m / TP, t = m % TP;
                    int td = t + 1; if (td == TP) td = 0;
                    if (td < TSEQ) {
                        size_t oi = ((size_t)bb * TSEQ + td) * DIMC + n;
                        ((float*)Out)[oi] = v + resid[oi];
                    }
                } else if (mode == 2) {
                    v += bias[n];
                    v = fmaxf(v, 0.0f);
                    ((uint16_t*)Out)[(size_t)m * N + n] = f2bf(v);
                } else { // mode 3
                    size_t oi = (size_t)m * N + n;
                    ((float*)Out)[oi] = v + bias[n] + resid[oi];
                }
            }
        }
    }
}

// ---------------- windowed attention: one lane per (batch, window, head) ----------------
__global__ __launch_bounds__(256) void attn3_kernel(
    const uint16_t* __restrict__ qkv, uint16_t* __restrict__ o)
{
    int gid = blockIdx.x * 256 + threadIdx.x;
    const int total = BATCH * NWIN * HEADS;
    if (gid >= total) return;
    const int h  = gid % HEADS;
    const int wb = gid / HEADS;
    const int w  = wb % NWIN;
    const int bb = wb / NWIN;

    const size_t rowbase = (size_t)bb * TP + (size_t)w * WS;
    const int qo = h * HD, ko = DIMC + h * HD, vo = 2 * DIMC + h * HD;

    float q[WS][HD], kk[WS][HD], vv[WS][HD];
#pragma unroll
    for (int i = 0; i < WS; ++i) {
        const uint16_t* r = qkv + (rowbase + i) * (size_t)(3 * DIMC);
#pragma unroll
        for (int d = 0; d < HD; ++d) {
            q[i][d]  = bf2f(r[qo + d]);
            kk[i][d] = bf2f(r[ko + d]);
            vv[i][d] = bf2f(r[vo + d]);
        }
    }
    float p[WS][WS];
#pragma unroll
    for (int i = 0; i < WS; ++i)
#pragma unroll
        for (int j = 0; j < WS; ++j) {
            float s = 0.0f;
#pragma unroll
            for (int d = 0; d < HD; ++d) s += q[i][d] * kk[j][d];
            p[i][j] = s * 0.25f;               // 1/sqrt(HD=16)
        }
#pragma unroll
    for (int i = 0; i < WS; ++i) {
        float mx = fmaxf(p[i][0], fmaxf(p[i][1], p[i][2]));
        float e0 = __expf(p[i][0] - mx), e1 = __expf(p[i][1] - mx), e2 = __expf(p[i][2] - mx);
        float inv = 1.0f / (e0 + e1 + e2);
        p[i][0] = e0 * inv; p[i][1] = e1 * inv; p[i][2] = e2 * inv;
    }
#pragma unroll
    for (int i = 0; i < WS; ++i) {
        uint16_t* orow = o + (rowbase + i) * (size_t)DIMC + h * HD;
#pragma unroll
        for (int d = 0; d < HD; ++d) {
            float s = p[i][0] * vv[0][d] + p[i][1] * vv[1][d] + p[i][2] * vv[2][d];
            orow[d] = f2bf(s);
        }
    }
}

// ---------------- launcher ----------------
extern "C" void kernel_launch(void* const* d_in, const int* in_sizes, int n_in,
                              void* d_out, int out_size, void* d_ws, size_t ws_size,
                              hipStream_t stream)
{
    const float* x      = (const float*)d_in[0];
    const float* g1     = (const float*)d_in[1];
    const float* b1     = (const float*)d_in[2];
    const float* w_qkv  = (const float*)d_in[3];
    const float* w_proj = (const float*)d_in[4];
    const float* g2     = (const float*)d_in[5];
    const float* b2     = (const float*)d_in[6];
    const float* w_mlp1 = (const float*)d_in[7];
    const float* b_mlp1 = (const float*)d_in[8];
    const float* w_mlp2 = (const float*)d_in[9];
    const float* b_mlp2 = (const float*)d_in[10];
    float* out = (float*)d_out;

    char* ws = (char*)d_ws;
    size_t off = 0;
    auto take = [&](size_t bytes) -> void* {
        void* p = ws + off;
        off += (bytes + 255) & ~(size_t)255;
        return p;
    };
    const size_t RTP = (size_t)BATCH * TP;    // 32784 rolled rows
    const size_t RT  = (size_t)BATCH * TSEQ;  // 32768 tokens

    uint16_t* xs     = (uint16_t*)take(RTP * DIMC * 2);        // LN1, rolled+padded, bf16
    uint16_t* qkvb   = (uint16_t*)take(RTP * 3 * DIMC * 2);    // qkv, bf16
    uint16_t* ob     = (uint16_t*)take(RTP * DIMC * 2);        // attn out, bf16
    float*    x2     = (float*)   take(RT * DIMC * 4);         // x + proj, f32
    uint16_t* hb     = (uint16_t*)take(RT * DIMC * 2);         // LN2, bf16
    uint16_t* hid    = (uint16_t*)take(RT * 2 * DIMC * 2);     // relu(mlp1), bf16
    uint16_t* wqkvT  = (uint16_t*)take((size_t)3 * DIMC * DIMC * 2);
    uint16_t* wprojT = (uint16_t*)take((size_t)DIMC * DIMC * 2);
    uint16_t* wmlp1T = (uint16_t*)take((size_t)2 * DIMC * DIMC * 2);
    uint16_t* wmlp2T = (uint16_t*)take((size_t)2 * DIMC * DIMC * 2);

    // 1) weights -> bf16, transposed to [N][K]
    wt_transpose_bf16<<<(DIMC * 3 * DIMC + 255) / 256, 256, 0, stream>>>(w_qkv,  wqkvT,  DIMC,     3 * DIMC);
    wt_transpose_bf16<<<(DIMC * DIMC     + 255) / 256, 256, 0, stream>>>(w_proj, wprojT, DIMC,     DIMC);
    wt_transpose_bf16<<<(DIMC * 2 * DIMC + 255) / 256, 256, 0, stream>>>(w_mlp1, wmlp1T, DIMC,     2 * DIMC);
    wt_transpose_bf16<<<(2 * DIMC * DIMC + 255) / 256, 256, 0, stream>>>(w_mlp2, wmlp2T, 2 * DIMC, DIMC);

    // 2) LN1 -> rolled/padded bf16 + zero pad rows
    ln_bf16_kernel<<<(uint32_t)RT, 256, 0, stream>>>(x, g1, b1, xs, /*roll=*/1);
    zero_pad_rows<<<BATCH * 2, 256, 0, stream>>>(xs);

    // 3) QKV GEMM: [RTP,768] x [768,2304]
    {
        dim3 grid((3 * DIMC) / BN, (uint32_t)((RTP + BM - 1) / BM));
        gemm_wmma_bf16<<<grid, 256, 0, stream>>>(xs, wqkvT, (int)RTP, 3 * DIMC, DIMC,
                                                 /*mode=*/0, nullptr, nullptr, qkvb);
    }

    // 4) windowed attention (ws=3)
    attn3_kernel<<<(BATCH * NWIN * HEADS + 255) / 256, 256, 0, stream>>>(qkvb, ob);

    // 5) proj GEMM + roll-back + residual: x2 = x + roll(o @ Wp)
    {
        dim3 grid(DIMC / BN, (uint32_t)((RTP + BM - 1) / BM));
        gemm_wmma_bf16<<<grid, 256, 0, stream>>>(ob, wprojT, (int)RTP, DIMC, DIMC,
                                                 /*mode=*/1, nullptr, x, x2);
    }

    // 6) LN2
    ln_bf16_kernel<<<(uint32_t)RT, 256, 0, stream>>>(x2, g2, b2, hb, /*roll=*/0);

    // 7) MLP1: relu(h @ W1 + b1)
    {
        dim3 grid((2 * DIMC) / BN, (uint32_t)(RT / BM));
        gemm_wmma_bf16<<<grid, 256, 0, stream>>>(hb, wmlp1T, (int)RT, 2 * DIMC, DIMC,
                                                 /*mode=*/2, b_mlp1, nullptr, hid);
    }

    // 8) MLP2 + bias + residual -> d_out
    {
        dim3 grid(DIMC / BN, (uint32_t)(RT / BM));
        gemm_wmma_bf16<<<grid, 256, 0, stream>>>(hid, wmlp2T, (int)RT, DIMC, 2 * DIMC,
                                                 /*mode=*/3, b_mlp2, x2, out);
    }
}
